// ResidualVectorQuantizer_86225763435036
// MI455X (gfx1250) — matrix operand order repaired and checked by reading, not compiled, not run on hardware
//
#include <hip/hip_runtime.h>
#include <hip/hip_bf16.h>

typedef __bf16 bf16_t;
typedef __attribute__((ext_vector_type(16))) __bf16 v16bf;
typedef __attribute__((ext_vector_type(8)))  __bf16 v8bf;
typedef __attribute__((ext_vector_type(8)))  float  v8f;

namespace {
constexpr int    Bn = 16, Dn = 512, Tn = 4096, Qn = 8, Kn = 1024;
constexpr int    ROWS_WG = 128;            // rows (b,t pairs) per workgroup
constexpr int    WAVES   = 8;              // wave32 waves per workgroup
constexpr int    THREADS = WAVES * 32;     // 256
constexpr int    RSTRIDE = Dn + 4;         // residual LDS row stride (floats) -> bank-safe
constexpr int    BSTRIDE = 520;            // B-tile row stride (bf16) = 1040 B, 16B-aligned rows
constexpr int    NBUF    = 3;              // B-tile ring depth
constexpr int    BTILE_BYTES = 16 * BSTRIDE * 2;              // 16,640 B per staged tile
constexpr size_t RES_BYTES   = (size_t)ROWS_WG * RSTRIDE * 4; // 264,192 B
constexpr size_t SMEM_BYTES  = RES_BYTES + (size_t)NBUF * BTILE_BYTES; // 314,112 B
constexpr size_t QUANT_N  = (size_t)Bn * Dn * Tn;   // 33,554,432
constexpr size_t IDX_N    = (size_t)Bn * Qn * Tn;   // 524,288
constexpr size_t LOSS_OFF = QUANT_N + IDX_N;
constexpr size_t CBH_ELEMS = (size_t)Qn * Kn * Dn;  // 4,194,304 bf16
constexpr size_t CBH_BYTES = CBH_ELEMS * sizeof(bf16_t);
}

// ---- prep: zero the loss accumulator slot in d_out ----
__global__ void rvq_zero_loss(float* out) {
    if (threadIdx.x == 0 && blockIdx.x == 0) out[LOSS_OFF] = 0.0f;
}

// ---- prep: fp32 codebooks -> bf16 copy (L2-resident GEMM operand) ----
__global__ __launch_bounds__(256) void rvq_cvt_cb(const float* __restrict__ cb,
                                                  bf16_t* __restrict__ cbh) {
    size_t i = (size_t)blockIdx.x * 256 + threadIdx.x;
    if (i < CBH_ELEMS) cbh[i] = (bf16_t)cb[i];
}

// ---- prep: per-code squared norms ||c||^2 ----
__global__ __launch_bounds__(256) void rvq_cnorm(const float* __restrict__ cb,
                                                 float* __restrict__ cn) {
    int code = blockIdx.x * 8 + (threadIdx.x >> 5);
    int lane = threadIdx.x & 31;
    const float* p = cb + (size_t)code * Dn;
    float s = 0.0f;
    #pragma unroll
    for (int u = 0; u < Dn / 32; ++u) { float v = p[u * 32 + lane]; s += v * v; }
    #pragma unroll
    for (int m = 1; m < 32; m <<= 1) s += __shfl_xor(s, m, 32);
    if (lane == 0) cn[code] = s;
}

// ---- async copy of one 16-code x 512-d bf16 tile (16 KB) into padded LDS ----
// Each wave moves its 2 KB share with 4 global_load_async_to_lds_b128 (ASYNCcnt).
__device__ __forceinline__ void stage_tile(const char* gtile, uint32_t lds_base,
                                           int wave, int lane) {
    #pragma unroll
    for (int i = 0; i < 4; ++i) {
        int off = wave * 2048 + i * 512 + lane * 16; // contiguous 16 KB in global
        int row = off >> 10;                         // code row (1 KB each)
        int col = off & 1023;
        uint32_t lds_off = lds_base + (uint32_t)(row * (BSTRIDE * 2) + col);
        uint64_t ga = (uint64_t)(uintptr_t)(gtile + off);
        asm volatile("global_load_async_to_lds_b128 %0, %1, off"
                     :: "v"(lds_off), "v"(ga) : "memory");
    }
}

__device__ __forceinline__ v16bf load_bfrag(const bf16_t* p) {
    v8bf lo = *(const v8bf*)p;
    v8bf hi = *(const v8bf*)(p + 8);
    return __builtin_shufflevector(lo, hi, 0,1,2,3,4,5,6,7,8,9,10,11,12,13,14,15);
}

// ---- main fused RVQ kernel ----
__global__ __launch_bounds__(THREADS) void rvq_main(const float*  __restrict__ x,
                                                    const float*  __restrict__ cb,
                                                    const bf16_t* __restrict__ cbh,
                                                    const float*  __restrict__ cnorm,
                                                    float* __restrict__ out) {
    extern __shared__ char smem[];
    float*  res   = (float*)smem;                       // [128][516] fp32 residual
    bf16_t* btile = (bf16_t*)(smem + RES_BYTES);        // 3 x padded 16-code tiles

    const int tid  = threadIdx.x;
    const int lane = tid & 31;
    const int wave = tid >> 5;
    const int row0 = blockIdx.x * ROWS_WG;              // global row = b*Tn + t
    const int b    = row0 / Tn;                         // 128 | 4096: no straddle
    const int t0   = row0 % Tn;
    const uint32_t bt0 = (uint32_t)(uintptr_t)btile;    // LDS byte offset of ring

    // ---- load x tile (B,D,T: row (b,t) strided by T) into LDS ----
    for (int i = tid; i < ROWS_WG * Dn; i += THREADS) {
        int tl = i & (ROWS_WG - 1);
        int d  = i >> 7;
        res[tl * RSTRIDE + d] = x[(size_t)b * Dn * Tn + (size_t)d * Tn + (t0 + tl)];
    }
    __syncthreads();

    const int half = lane >> 4;      // WMMA 16-bit operand K-subset selector
    const int mloc = lane & 15;      // A row / B column within tile
    const int rowBase = wave * 16;   // this wave's 16 rows (local)

    float loss_sq = 0.0f;

    for (int q = 0; q < Qn; ++q) {
        const char* cbq = (const char*)(cbh + (size_t)q * Kn * Dn);

        // ---- build 16 bf16 A-fragments (all of D) for this wave's rows ----
        v16bf A[16];
        {
            const float* rrow = &res[(rowBase + mloc) * RSTRIDE];
            #pragma unroll
            for (int s = 0; s < 16; ++s) {
                const int d0 = s * 32;
                v16bf a;
                #pragma unroll
                for (int j = 0; j < 8; ++j) a[j]     = (bf16_t)rrow[d0 + half * 16 + j];
                #pragma unroll
                for (int j = 0; j < 8; ++j) a[8 + j] = (bf16_t)rrow[d0 + half * 16 + 8 + j];
                A[s] = a;
            }
        }

        float bestS[8];
        int   bestI[8];
        #pragma unroll
        for (int j = 0; j < 8; ++j) { bestS[j] = 3.4e38f; bestI[j] = 0; }

        // ---- pipeline prologue: stage column-tile 0 ----
        stage_tile(cbq, bt0, wave, lane);

        for (int ct = 0; ct < Kn / 16; ++ct) {
            // issue next tile (overwrites buffer computed 2 iters ago; the
            // per-iter barrier below makes that provably safe), then wait for
            // this iter's tile and sync all waves' shares.
            if (ct < Kn / 16 - 1) {
                stage_tile(cbq + (ct + 1) * 16384,
                           bt0 + (uint32_t)(((ct + 1) % NBUF) * BTILE_BYTES),
                           wave, lane);
                asm volatile("s_wait_asynccnt 0x4" ::: "memory");
            } else {
                asm volatile("s_wait_asynccnt 0x0" ::: "memory");
            }
            __syncthreads();

            const bf16_t* bb   = (const bf16_t*)((const char*)btile + (ct % NBUF) * BTILE_BYTES);
            const bf16_t* rowp = bb + mloc * BSTRIDE + half * 16;

            v8f c = {};
            #pragma unroll
            for (int s = 0; s < 16; ++s) {
                v16bf bcur = load_bfrag(rowp + s * 32);
                c = __builtin_amdgcn_wmma_f32_16x16x32_bf16(
                        false, A[s], false, bcur, (short)0, c, false, false);
            }

            const int code = ct * 16 + mloc;
            const float cn = cnorm[q * Kn + code];
            #pragma unroll
            for (int j = 0; j < 8; ++j) {
                float sc = __builtin_fmaf(-2.0f, c[j], cn); // ||c||^2 - 2 r.c
                if (sc < bestS[j]) { bestS[j] = sc; bestI[j] = code; }
            }
        }

        // ---- argmin reduce across the 16 lanes of each half ----
        #pragma unroll
        for (int j = 0; j < 8; ++j) {
            float s = bestS[j];
            int   id = bestI[j];
            #pragma unroll
            for (int m = 1; m < 16; m <<= 1) {
                float so = __shfl_xor(s, m, 32);
                int   io = __shfl_xor(id, m, 32);
                if (so < s || (so == s && io < id)) { s = so; id = io; }
            }
            bestI[j] = id;
        }

        // ---- write indices (int32): lanes 0/16 own rows j / 8+j ----
        if (mloc == 0) {
            #pragma unroll
            for (int j = 0; j < 8; ++j) {
                int t = t0 + rowBase + half * 8 + j;
                ((int*)out)[QUANT_N + (size_t)b * Qn * Tn + (size_t)q * Tn + t] = bestI[j];
            }
        }

        // ---- gather chosen fp32 codes (L2), update residual, loss ----
        #pragma unroll
        for (int j = 0; j < 8; ++j) {
            int idLo = __shfl(bestI[j], 0, 32);   // row rowBase + j
            int idHi = __shfl(bestI[j], 16, 32);  // row rowBase + 8 + j
            const float* cLo = cb + ((size_t)q * Kn + idLo) * Dn;
            const float* cHi = cb + ((size_t)q * Kn + idHi) * Dn;
            float* rLo = &res[(rowBase + j) * RSTRIDE];
            float* rHi = &res[(rowBase + 8 + j) * RSTRIDE];
            #pragma unroll
            for (int u = 0; u < Dn / 32; ++u) {
                int d = u * 32 + lane;
                float nLo = rLo[d] - cLo[d];
                float nHi = rHi[d] - cHi[d];
                rLo[d] = nLo;
                rHi[d] = nHi;
                loss_sq += nLo * nLo + nHi * nHi;
            }
        }
        __syncthreads();  // stage boundary: protects B ring reuse + res reads
    }

    // ---- commitment loss: 0.25 * sum_stages mean(new_residual^2) ----
    #pragma unroll
    for (int m = 1; m < 32; m <<= 1) loss_sq += __shfl_xor(loss_sq, m, 32);
    if (lane == 0)
        atomicAdd(&out[LOSS_OFF], loss_sq * (0.25f / (float)QUANT_N));

    // ---- quantized = x - residual_final, back to (B,D,T) layout ----
    for (int i = tid; i < ROWS_WG * Dn; i += THREADS) {
        int tl = i & (ROWS_WG - 1);
        int d  = i >> 7;
        size_t g = (size_t)b * Dn * Tn + (size_t)d * Tn + (t0 + tl);
        out[g] = x[g] - res[tl * RSTRIDE + d];
    }
}

extern "C" void kernel_launch(void* const* d_in, const int* in_sizes, int n_in,
                              void* d_out, int out_size, void* d_ws, size_t ws_size,
                              hipStream_t stream) {
    const float* x  = (const float*)d_in[0];
    const float* cb = (const float*)d_in[1];
    float* out = (float*)d_out;

    bf16_t* cbh   = (bf16_t*)d_ws;
    float*  cnorm = (float*)((char*)d_ws + CBH_BYTES);

    (void)in_sizes; (void)n_in; (void)out_size; (void)ws_size;

    hipFuncSetAttribute((const void*)rvq_main,
                        hipFuncAttributeMaxDynamicSharedMemorySize,
                        (int)SMEM_BYTES);

    rvq_zero_loss<<<1, 32, 0, stream>>>(out);
    rvq_cvt_cb<<<(int)((CBH_ELEMS + 255) / 256), 256, 0, stream>>>(cb, cbh);
    rvq_cnorm<<<(Qn * Kn) / 8, 256, 0, stream>>>(cb, cnorm);

    const int nblocks = (Bn * Tn) / ROWS_WG; // 512
    rvq_main<<<nblocks, THREADS, SMEM_BYTES, stream>>>(x, cb, cbh, cnorm, out);
}
